// AttnRNNDecoder_317827580215
// MI455X (gfx1250) — compile-verified, matrix operand order checked by reading, not run on hardware
//
#include <hip/hip_runtime.h>
#include <hip/hip_bf16.h>
#include <stdint.h>

// Problem constants (from reference)
#define T_STEPS 32
#define Bsz     64
#define Ssz     64
#define Esz     512
#define Hsz     1024
#define Vsz     32000
#define IN0     1537           // E + 1 + H
#define IN0P    1568           // padded to multiple of 32 for WMMA K
#define NSPEC   5              // special_emb rows
#define G3H     3072           // 3*H

typedef __hip_bfloat16 bf16;
typedef __attribute__((ext_vector_type(16))) __bf16 v16bf;
typedef __attribute__((ext_vector_type(8)))  float  v8f;

// ---------------------------------------------------------------------------
// WMMA fragment load: row-major [rows x K] bf16 source, 16x32 fragment.
// Lane L<16  : row = L,    K = {k..k+7, k+16..k+23}
// Lane L>=16 : row = L-16, K = {k+8..k+15, k+24..k+31}
// Works for global or LDS-backed pointers (ds_load_b128 when addrspace known).
// ---------------------------------------------------------------------------
__device__ __forceinline__ v16bf load_frag(const bf16* __restrict__ p, int ld, int lane) {
    const bf16* base = p + (long)(lane & 15) * ld + ((lane >> 4) << 3);
    union { v16bf v; uint4 u[2]; } t;
    t.u[0] = *reinterpret_cast<const uint4*>(base);        // K octet 0
    t.u[1] = *reinterpret_cast<const uint4*>(base + 16);   // K octet +16
    return t.v;
}

// Async global -> LDS copy, 16B per lane. lds_off is a byte offset relative to
// the wave's LDS base.
__device__ __forceinline__ void async_load_b128(uint32_t lds_off, const void* gaddr) {
    asm volatile("global_load_async_to_lds_b128 %0, %1, off"
                 :: "v"(lds_off), "v"(gaddr) : "memory");
}

// ---------------------------------------------------------------------------
// Direct-global bf16 GEMM: C[M,N](f32) = A[M,K] * B[N,K]^T (+bias). Wave tile
// 32x64, no LDS. Used for the small latency-bound per-step GEMMs (operands L2-
// resident). M%32==0, N%64==0, K%32==0.
// ---------------------------------------------------------------------------
__global__ void k_gemm_bf16(const bf16* __restrict__ A, int lda,
                            const bf16* __restrict__ B, int ldb,
                            float* __restrict__ C, long ldc,
                            const float* __restrict__ bias,
                            int M, int N, int K) {
    const int lane   = threadIdx.x & 31;
    const int wid    = blockIdx.x * 4 + (threadIdx.x >> 5);
    const int ntiles = N >> 6;
    const int mtiles = M >> 5;
    if (wid >= ntiles * mtiles) return;
    const int row0 = (wid / ntiles) << 5;
    const int col0 = (wid % ntiles) << 6;

    const bf16* Ar0 = A + (long)row0 * lda;
    const bf16* Ar1 = Ar0 + (long)16 * lda;
    const bf16* Bc0 = B + (long)col0 * ldb;
    const bf16* Bc1 = Bc0 + (long)16 * ldb;
    const bf16* Bc2 = Bc0 + (long)32 * ldb;
    const bf16* Bc3 = Bc0 + (long)48 * ldb;

    v8f acc00 = {}, acc01 = {}, acc02 = {}, acc03 = {};
    v8f acc10 = {}, acc11 = {}, acc12 = {}, acc13 = {};

    for (int k = 0; k < K; k += 32) {
        v16bf a0 = load_frag(Ar0 + k, lda, lane);
        v16bf a1 = load_frag(Ar1 + k, lda, lane);
        v16bf b0 = load_frag(Bc0 + k, ldb, lane);
        v16bf b1 = load_frag(Bc1 + k, ldb, lane);
        v16bf b2 = load_frag(Bc2 + k, ldb, lane);
        v16bf b3 = load_frag(Bc3 + k, ldb, lane);
        acc00 = __builtin_amdgcn_wmma_f32_16x16x32_bf16(false, a0, false, b0, (short)0, acc00, false, false);
        acc01 = __builtin_amdgcn_wmma_f32_16x16x32_bf16(false, a0, false, b1, (short)0, acc01, false, false);
        acc02 = __builtin_amdgcn_wmma_f32_16x16x32_bf16(false, a0, false, b2, (short)0, acc02, false, false);
        acc03 = __builtin_amdgcn_wmma_f32_16x16x32_bf16(false, a0, false, b3, (short)0, acc03, false, false);
        acc10 = __builtin_amdgcn_wmma_f32_16x16x32_bf16(false, a1, false, b0, (short)0, acc10, false, false);
        acc11 = __builtin_amdgcn_wmma_f32_16x16x32_bf16(false, a1, false, b1, (short)0, acc11, false, false);
        acc12 = __builtin_amdgcn_wmma_f32_16x16x32_bf16(false, a1, false, b2, (short)0, acc12, false, false);
        acc13 = __builtin_amdgcn_wmma_f32_16x16x32_bf16(false, a1, false, b3, (short)0, acc13, false, false);
    }

    const int n_lo  = col0 + (lane & 15);
    const int mbase = row0 + ((lane >> 4) << 3);
    v8f* accs0[4] = { &acc00, &acc01, &acc02, &acc03 };
    v8f* accs1[4] = { &acc10, &acc11, &acc12, &acc13 };
#pragma unroll
    for (int j = 0; j < 4; ++j) {
        const int n  = n_lo + j * 16;
        const float bv = bias ? bias[n] : 0.0f;
        v8f a = *accs0[j];
        v8f b = *accs1[j];
#pragma unroll
        for (int i = 0; i < 8; ++i) {
            C[(long)(mbase + i)      * ldc + n] = a[i] + bv;
            C[(long)(mbase + 16 + i) * ldc + n] = b[i] + bv;
        }
    }
}

// ---------------------------------------------------------------------------
// Generator GEMM: block tile 64(M) x 256(N), K-chunk 64, double-buffered LDS
// fed by async global->LDS b128 copies (ASYNCcnt), 8 waves in a 2x4 grid each
// computing a 32x64 wave tile (16 WMMAs per chunk) out of LDS.
// LDS per buffer: A 64x64 bf16 (8KB) + B 256x64 bf16 (32KB) = 40KB; 80KB total.
// Requires M%64==0, N%256==0, K%64==0 (generator: 2048, 32000, 1024).
// ---------------------------------------------------------------------------
#define GEN_BM 64
#define GEN_BN 256
#define GEN_BK 64
#define GEN_BUF 40960u

__global__ void __launch_bounds__(256) k_gemm_gen(const bf16* __restrict__ A,
                                                  const bf16* __restrict__ B,
                                                  float* __restrict__ C,
                                                  const float* __restrict__ bias,
                                                  int M, int N, int K) {
    extern __shared__ char smem[];
    const int tid  = threadIdx.x;
    const int lane = tid & 31;
    const int w    = tid >> 5;     // 0..7
    const int wm   = w >> 2;       // 0..1  (M sub-tile)
    const int wn   = w & 3;        // 0..3  (N sub-tile)

    const int nblk = N / GEN_BN;
    const int m0 = (blockIdx.x / nblk) * GEN_BM;
    const int n0 = (blockIdx.x % nblk) * GEN_BN;

    // per-thread staging coords: 8 threads per 64-col row, 32 rows per issue
    const int lr = tid >> 3;          // 0..31
    const int lc = (tid & 7) * 8;     // element offset 0..56

    const uint32_t aOff[2] = { 0u, GEN_BUF };
    const uint32_t bOff[2] = { 8192u, GEN_BUF + 8192u };

    auto issue = [&](int kc, int bufi) {
#pragma unroll
        for (int j = 0; j < 2; ++j) {            // A tile: 64 rows
            int r = lr + j * 32;
            async_load_b128(aOff[bufi] + (uint32_t)(r * GEN_BK + lc) * 2u,
                            A + (size_t)(m0 + r) * K + kc + lc);
        }
#pragma unroll
        for (int j = 0; j < 8; ++j) {            // B tile: 256 rows
            int r = lr + j * 32;
            async_load_b128(bOff[bufi] + (uint32_t)(r * GEN_BK + lc) * 2u,
                            B + (size_t)(n0 + r) * K + kc + lc);
        }
    };

    v8f acc00 = {}, acc01 = {}, acc02 = {}, acc03 = {};
    v8f acc10 = {}, acc11 = {}, acc12 = {}, acc13 = {};

    issue(0, 0);
    const int nchunks = K / GEN_BK;
    for (int i = 0; i < nchunks; ++i) {
        if (i + 1 < nchunks) {
            issue((i + 1) * GEN_BK, (i + 1) & 1);
            asm volatile("s_wait_asynccnt 0xa" ::: "memory");  // chunk i done (in-order)
        } else {
            asm volatile("s_wait_asynccnt 0x0" ::: "memory");
        }
        __syncthreads();   // all waves' async slices visible in LDS

        const bf16* sA = (const bf16*)(smem + aOff[i & 1]);
        const bf16* sB = (const bf16*)(smem + bOff[i & 1]);
#pragma unroll
        for (int ks = 0; ks < GEN_BK; ks += 32) {
            v16bf a0 = load_frag(sA + (wm * 32)      * GEN_BK + ks, GEN_BK, lane);
            v16bf a1 = load_frag(sA + (wm * 32 + 16) * GEN_BK + ks, GEN_BK, lane);
            v16bf b0 = load_frag(sB + (wn * 64)      * GEN_BK + ks, GEN_BK, lane);
            v16bf b1 = load_frag(sB + (wn * 64 + 16) * GEN_BK + ks, GEN_BK, lane);
            v16bf b2 = load_frag(sB + (wn * 64 + 32) * GEN_BK + ks, GEN_BK, lane);
            v16bf b3 = load_frag(sB + (wn * 64 + 48) * GEN_BK + ks, GEN_BK, lane);
            acc00 = __builtin_amdgcn_wmma_f32_16x16x32_bf16(false, a0, false, b0, (short)0, acc00, false, false);
            acc01 = __builtin_amdgcn_wmma_f32_16x16x32_bf16(false, a0, false, b1, (short)0, acc01, false, false);
            acc02 = __builtin_amdgcn_wmma_f32_16x16x32_bf16(false, a0, false, b2, (short)0, acc02, false, false);
            acc03 = __builtin_amdgcn_wmma_f32_16x16x32_bf16(false, a0, false, b3, (short)0, acc03, false, false);
            acc10 = __builtin_amdgcn_wmma_f32_16x16x32_bf16(false, a1, false, b0, (short)0, acc10, false, false);
            acc11 = __builtin_amdgcn_wmma_f32_16x16x32_bf16(false, a1, false, b1, (short)0, acc11, false, false);
            acc12 = __builtin_amdgcn_wmma_f32_16x16x32_bf16(false, a1, false, b2, (short)0, acc12, false, false);
            acc13 = __builtin_amdgcn_wmma_f32_16x16x32_bf16(false, a1, false, b3, (short)0, acc13, false, false);
        }
        __syncthreads();   // everyone done reading buf[i&1] before it is refilled
    }

    const int row0  = m0 + wm * 32;
    const int col0  = n0 + wn * 64;
    const int n_lo  = col0 + (lane & 15);
    const int mbase = row0 + ((lane >> 4) << 3);
    v8f* accs0[4] = { &acc00, &acc01, &acc02, &acc03 };
    v8f* accs1[4] = { &acc10, &acc11, &acc12, &acc13 };
#pragma unroll
    for (int j = 0; j < 4; ++j) {
        const int n  = n_lo + j * 16;
        const float bv = bias ? bias[n] : 0.0f;
        v8f a = *accs0[j];
        v8f b = *accs1[j];
#pragma unroll
        for (int i = 0; i < 8; ++i) {
            C[(long)(mbase + i)      * N + n] = a[i] + bv;
            C[(long)(mbase + 16 + i) * N + n] = b[i] + bv;
        }
    }
}

// ---------------------------------------------------------------------------
// Elementwise / helper kernels
// ---------------------------------------------------------------------------
__global__ void k_f32_to_bf16(const float* __restrict__ s, bf16* __restrict__ d, long n) {
    long i = (long)blockIdx.x * blockDim.x + threadIdx.x;
    long st = (long)gridDim.x * blockDim.x;
    for (; i < n; i += st) d[i] = __float2bfloat16(s[i]);
}

__global__ void k_convert_pad(const float* __restrict__ s, bf16* __restrict__ d,
                              int rows, int ks, int kd) {
    long n = (long)rows * kd;
    long i = (long)blockIdx.x * blockDim.x + threadIdx.x;
    long st = (long)gridDim.x * blockDim.x;
    for (; i < n; i += st) {
        int r = (int)(i / kd), c = (int)(i % kd);
        d[i] = __float2bfloat16(c < ks ? s[(long)r * ks + c] : 0.0f);
    }
}

__global__ void k_embed_concat(const int* __restrict__ ids, const float* __restrict__ lc,
                               const float* __restrict__ out_state,
                               const float* __restrict__ word_emb,
                               const float* __restrict__ special_emb,
                               bf16* __restrict__ inp) {
    int idx = blockIdx.x * blockDim.x + threadIdx.x;
    if (idx >= Bsz * IN0P) return;
    int b = idx / IN0P, j = idx % IN0P;
    float v;
    if (j < Esz) {
        int id = ids[b];
        int sid = (id < NSPEC) ? id : 0;
        v = word_emb[(long)id * Esz + j] + special_emb[(long)sid * Esz + j];
    } else if (j == Esz) {
        v = lc[b];
    } else if (j < IN0) {
        v = out_state[b * Hsz + (j - Esz - 1)];
    } else {
        v = 0.0f;
    }
    inp[idx] = __float2bfloat16(v);
}

__global__ void k_gru_elem(const float* __restrict__ gi, const float* __restrict__ gh,
                           const float* __restrict__ bih, const float* __restrict__ bhh,
                           const float* __restrict__ hprev,
                           float* __restrict__ hnew, bf16* __restrict__ hnewb) {
    int idx = blockIdx.x * blockDim.x + threadIdx.x;   // B*H
    if (idx >= Bsz * Hsz) return;
    int b = idx >> 10, j = idx & (Hsz - 1);
    const float* gib = gi + (long)b * G3H;
    const float* ghb = gh + (long)b * G3H;
    float ir = gib[j]           + bih[j],           hr = ghb[j]           + bhh[j];
    float iz = gib[Hsz + j]     + bih[Hsz + j],     hz = ghb[Hsz + j]     + bhh[Hsz + j];
    float in = gib[2 * Hsz + j] + bih[2 * Hsz + j], hn = ghb[2 * Hsz + j] + bhh[2 * Hsz + j];
    float r = 1.0f / (1.0f + __expf(-(ir + hr)));
    float z = 1.0f / (1.0f + __expf(-(iz + hz)));
    float n = tanhf(in + r * hn);
    float h = (1.0f - z) * n + z * hprev[idx];
    hnew[idx]  = h;
    hnewb[idx] = __float2bfloat16(h);
}

__global__ void k_attn_score(const float* __restrict__ context, const float* __restrict__ q,
                             const uint8_t* __restrict__ mask, float* __restrict__ align) {
    int bs = blockIdx.x;               // B*S blocks
    int b = bs >> 6, s = bs & (Ssz - 1);
    const float* c  = context + ((long)s * Bsz + b) * Hsz;
    const float* qb = q + (long)b * Hsz;
    float sum = 0.0f;
    for (int h = threadIdx.x; h < Hsz; h += 128) sum += c[h] * qb[h];
    __shared__ float red[128];
    red[threadIdx.x] = sum; __syncthreads();
    for (int off = 64; off > 0; off >>= 1) {
        if (threadIdx.x < off) red[threadIdx.x] += red[threadIdx.x + off];
        __syncthreads();
    }
    if (threadIdx.x == 0) {
        float a = red[0];
        if (!mask[b * Ssz + s]) a = -1e9f;
        align[b * Ssz + s] = a;
    }
}

__global__ void k_softmax64(const float* __restrict__ align, float* __restrict__ w) {
    int b = blockIdx.x, s = threadIdx.x;   // 64 threads
    __shared__ float sh[Ssz];
    float v = align[b * Ssz + s];
    sh[s] = v; __syncthreads();
    float mx = -3.4e38f;
    for (int i = 0; i < Ssz; ++i) mx = fmaxf(mx, sh[i]);
    float e = __expf(v - mx);
    __syncthreads(); sh[s] = e; __syncthreads();
    float sum = 0.0f;
    for (int i = 0; i < Ssz; ++i) sum += sh[i];
    w[b * Ssz + s] = e / sum;
}

__global__ void k_cvec(const float* __restrict__ w, const float* __restrict__ context,
                       bf16* __restrict__ cvecb) {
    int idx = blockIdx.x * blockDim.x + threadIdx.x;
    if (idx >= Bsz * Hsz) return;
    int b = idx >> 10, h = idx & (Hsz - 1);
    float sum = 0.0f;
    for (int s = 0; s < Ssz; ++s)
        sum += w[b * Ssz + s] * context[((long)s * Bsz + b) * Hsz + h];
    cvecb[idx] = __float2bfloat16(sum);
}

__global__ void k_tanh_out(const float* __restrict__ t1, const float* __restrict__ t2,
                           float* __restrict__ out_state, bf16* __restrict__ out_all_b) {
    int idx = blockIdx.x * blockDim.x + threadIdx.x;
    if (idx >= Bsz * Hsz) return;
    float v = tanhf(t1[idx] + t2[idx]);
    out_state[idx] = v;
    out_all_b[idx] = __float2bfloat16(v);
}

__global__ void k_log_softmax(float* __restrict__ scores) {
    float* p = scores + (long)blockIdx.x * Vsz;
    __shared__ float red[256];
    int tid = threadIdx.x;
    float m = -3.4e38f;
    for (int v = tid; v < Vsz; v += 256) m = fmaxf(m, p[v]);
    red[tid] = m; __syncthreads();
    for (int off = 128; off > 0; off >>= 1) {
        if (tid < off) red[tid] = fmaxf(red[tid], red[tid + off]);
        __syncthreads();
    }
    float mx = red[0]; __syncthreads();
    float s = 0.0f;
    for (int v = tid; v < Vsz; v += 256) s += __expf(p[v] - mx);
    red[tid] = s; __syncthreads();
    for (int off = 128; off > 0; off >>= 1) {
        if (tid < off) red[tid] += red[tid + off];
        __syncthreads();
    }
    float lse = mx + logf(red[0]);
    for (int v = tid; v < Vsz; v += 256) p[v] -= lse;
}

// ---------------------------------------------------------------------------
extern "C" void kernel_launch(void* const* d_in, const int* in_sizes, int n_in,
                              void* d_out, int out_size, void* d_ws, size_t ws_size,
                              hipStream_t stream) {
    const int*     ids        = (const int*)    d_in[0];
    const float*   lc         = (const float*)  d_in[1];
    const float*   hidden     = (const float*)  d_in[2];
    const float*   context    = (const float*)  d_in[3];
    const uint8_t* mask       = (const uint8_t*)d_in[4];
    const float*   prev_out   = (const float*)  d_in[5];
    const float*   word_emb   = (const float*)  d_in[6];
    const float*   spec_emb   = (const float*)  d_in[7];
    const float*   g0_Wih     = (const float*)  d_in[8];
    const float*   g0_Whh     = (const float*)  d_in[9];
    const float*   g0_bih     = (const float*)  d_in[10];
    const float*   g0_bhh     = (const float*)  d_in[11];
    const float*   g1_Wih     = (const float*)  d_in[12];
    const float*   g1_Whh     = (const float*)  d_in[13];
    const float*   g1_bih     = (const float*)  d_in[14];
    const float*   g1_bhh     = (const float*)  d_in[15];
    const float*   W_align    = (const float*)  d_in[16];
    const float*   W_ctx      = (const float*)  d_in[17];
    const float*   W_query    = (const float*)  d_in[18];
    const float*   W_gen      = (const float*)  d_in[19];
    const float*   b_gen      = (const float*)  d_in[20];

    float* out_f = (float*)d_out;  // [scores T*B*V | h_fin 2*B*H | out_fin B*H]

    char* wsp = (char*)d_ws;
    auto alloc = [&](size_t bytes) -> void* {
        void* p = (void*)wsp;
        wsp += (bytes + 255) & ~(size_t)255;
        return p;
    };
    bf16*  Wih0b   = (bf16*) alloc((size_t)G3H * IN0P * 2);
    bf16*  Whh0b   = (bf16*) alloc((size_t)G3H * Hsz * 2);
    bf16*  Wih1b   = (bf16*) alloc((size_t)G3H * Hsz * 2);
    bf16*  Whh1b   = (bf16*) alloc((size_t)G3H * Hsz * 2);
    bf16*  Walignb = (bf16*) alloc((size_t)Hsz * Hsz * 2);
    bf16*  Wctxb   = (bf16*) alloc((size_t)Hsz * Hsz * 2);
    bf16*  Wqueryb = (bf16*) alloc((size_t)Hsz * Hsz * 2);
    bf16*  Wgenb   = (bf16*) alloc((size_t)Vsz * Hsz * 2);
    float* h_state = (float*)alloc((size_t)2 * Bsz * Hsz * 4);
    bf16*  h01b    = (bf16*) alloc((size_t)2 * Bsz * Hsz * 2);
    float* out_st  = (float*)alloc((size_t)Bsz * Hsz * 4);
    bf16*  inp_b   = (bf16*) alloc((size_t)Bsz * IN0P * 2);
    float* gi      = (float*)alloc((size_t)Bsz * G3H * 4);
    float* gh      = (float*)alloc((size_t)Bsz * G3H * 4);
    float* qbuf    = (float*)alloc((size_t)Bsz * Hsz * 4);
    float* align   = (float*)alloc((size_t)Bsz * Ssz * 4);
    float* wts     = (float*)alloc((size_t)Bsz * Ssz * 4);
    bf16*  cvecb   = (bf16*) alloc((size_t)Bsz * Hsz * 2);
    float* t1      = (float*)alloc((size_t)Bsz * Hsz * 4);
    float* t2      = (float*)alloc((size_t)Bsz * Hsz * 4);
    bf16*  out_all = (bf16*) alloc((size_t)T_STEPS * Bsz * Hsz * 2);
    bf16*  h0b = h01b;
    bf16*  h1b = h01b + (size_t)Bsz * Hsz;

    auto cvt = [&](const float* s, bf16* d, long n) {
        int blocks = (int)((n + 255) / 256); if (blocks > 4096) blocks = 4096;
        k_f32_to_bf16<<<blocks, 256, 0, stream>>>(s, d, n);
    };
    auto gemm = [&](const bf16* A, int lda, const bf16* Bm, int ldb,
                    float* C, long ldc, const float* bias, int M, int N, int K) {
        int tiles = (M / 32) * (N / 64);
        k_gemm_bf16<<<(tiles + 3) / 4, 128, 0, stream>>>(A, lda, Bm, ldb, C, ldc, bias, M, N, K);
    };

    // ---- per-launch weight conversion to bf16 (~100MB -> L2-resident) ----
    k_convert_pad<<<4096, 256, 0, stream>>>(g0_Wih, Wih0b, G3H, IN0, IN0P);
    cvt(g0_Whh,  Whh0b,   (long)G3H * Hsz);
    cvt(g1_Wih,  Wih1b,   (long)G3H * Hsz);
    cvt(g1_Whh,  Whh1b,   (long)G3H * Hsz);
    cvt(W_align, Walignb, (long)Hsz * Hsz);
    cvt(W_ctx,   Wctxb,   (long)Hsz * Hsz);
    cvt(W_query, Wqueryb, (long)Hsz * Hsz);
    cvt(W_gen,   Wgenb,   (long)Vsz * Hsz);

    // ---- init recurrent state ----
    hipMemcpyAsync(h_state, hidden, (size_t)2 * Bsz * Hsz * 4, hipMemcpyDeviceToDevice, stream);
    hipMemcpyAsync(out_st,  prev_out, (size_t)Bsz * Hsz * 4,   hipMemcpyDeviceToDevice, stream);
    cvt(hidden, h01b, (long)2 * Bsz * Hsz);

    const int BH_BLOCKS = (Bsz * Hsz + 255) / 256;

    // ---- sequential decoder loop ----
    for (int t = 0; t < T_STEPS; ++t) {
        k_embed_concat<<<(Bsz * IN0P + 255) / 256, 256, 0, stream>>>(
            ids + t * Bsz, lc + t * Bsz, out_st, word_emb, spec_emb, inp_b);

        gemm(inp_b, IN0P, Wih0b, IN0P, gi, G3H, nullptr, Bsz, G3H, IN0P);
        gemm(h0b,   Hsz,  Whh0b, Hsz,  gh, G3H, nullptr, Bsz, G3H, Hsz);
        k_gru_elem<<<BH_BLOCKS, 256, 0, stream>>>(gi, gh, g0_bih, g0_bhh,
                                                  h_state, h_state, h0b);
        gemm(h0b, Hsz, Wih1b, Hsz, gi, G3H, nullptr, Bsz, G3H, Hsz);
        gemm(h1b, Hsz, Whh1b, Hsz, gh, G3H, nullptr, Bsz, G3H, Hsz);
        k_gru_elem<<<BH_BLOCKS, 256, 0, stream>>>(gi, gh, g1_bih, g1_bhh,
                                                  h_state + Bsz * Hsz,
                                                  h_state + Bsz * Hsz, h1b);
        gemm(h1b, Hsz, Walignb, Hsz, qbuf, Hsz, nullptr, Bsz, Hsz, Hsz);
        k_attn_score<<<Bsz * Ssz, 128, 0, stream>>>(context, qbuf, mask, align);
        k_softmax64<<<Bsz, Ssz, 0, stream>>>(align, wts);
        k_cvec<<<BH_BLOCKS, 256, 0, stream>>>(wts, context, cvecb);
        gemm(cvecb, Hsz, Wctxb,   Hsz, t1, Hsz, nullptr, Bsz, Hsz, Hsz);
        gemm(h1b,   Hsz, Wqueryb, Hsz, t2, Hsz, nullptr, Bsz, Hsz, Hsz);
        k_tanh_out<<<BH_BLOCKS, 256, 0, stream>>>(t1, t2, out_st,
                                                  out_all + (long)t * Bsz * Hsz);
    }

    // ---- batched generator: LDS-staged async-copy GEMM + bias, then log_softmax ----
    {
        int blocks = (T_STEPS * Bsz / GEN_BM) * (Vsz / GEN_BN);   // 32 * 125
        k_gemm_gen<<<blocks, 256, 2 * GEN_BUF, stream>>>(
            out_all, Wgenb, out_f, b_gen, T_STEPS * Bsz, Vsz, Hsz);
    }
    k_log_softmax<<<T_STEPS * Bsz, 256, 0, stream>>>(out_f);

    // ---- final states ----
    float* h_fin_dst   = out_f + (long)T_STEPS * Bsz * Vsz;
    float* out_fin_dst = h_fin_dst + 2 * Bsz * Hsz;
    hipMemcpyAsync(h_fin_dst,   h_state, (size_t)2 * Bsz * Hsz * 4, hipMemcpyDeviceToDevice, stream);
    hipMemcpyAsync(out_fin_dst, out_st,  (size_t)Bsz * Hsz * 4,     hipMemcpyDeviceToDevice, stream);
}
